// Encoder_87196426043432
// MI455X (gfx1250) — matrix operand order repaired and checked by reading, not compiled, not run on hardware
//
#include <hip/hip_runtime.h>

// ---------------------------------------------------------------------------
// Types for CDNA5 WMMA (wave32): v16bf = A/B fragment (8 VGPRs), v8f = C/D.
// ---------------------------------------------------------------------------
typedef __bf16 bf16;
typedef __attribute__((ext_vector_type(16))) __bf16 v16bf;
typedef __attribute__((ext_vector_type(8)))  __bf16 bf16x8;
typedef __attribute__((ext_vector_type(8)))  float   v8f;

#define T_NL   4096
#define L_NAME 32
#define NB     1024   // combined name batch (NF + NM)
#define E_DIM  1024
#define H_DIM  512
#define G4     2048   // 4*H
#define NFLD   512

__device__ __forceinline__ bf16 f2bf(float f) {
  unsigned u = __builtin_bit_cast(unsigned, f);
  u += 0x7FFFu + ((u >> 16) & 1u);              // round-to-nearest-even
  unsigned short s = (unsigned short)(u >> 16);
  return __builtin_bit_cast(bf16, s);
}
__device__ __forceinline__ float sigm(float x) { return 1.0f / (1.0f + __expf(-x)); }

// ---------------------------------------------------------------------------
// f32 -> bf16 conversion (weights)
// ---------------------------------------------------------------------------
__global__ void k_f32_to_bf16(const float* __restrict__ src, bf16* __restrict__ dst, int n) {
  int i = blockIdx.x * blockDim.x + threadIdx.x;
  if (i < n) dst[i] = f2bf(src[i]);
}

// ---------------------------------------------------------------------------
// Embedding gathers (-> bf16)
// ---------------------------------------------------------------------------
__global__ void k_gather_rows(const int* __restrict__ ids, const float* __restrict__ tab,
                              bf16* __restrict__ dst, int rows, int cols) {
  int i = blockIdx.x * blockDim.x + threadIdx.x;
  if (i >= rows * cols) return;
  int r = i / cols, c = i - r * cols;
  dst[i] = f2bf(tab[(size_t)ids[r] * cols + c]);
}

// names: [t][b][e], b<512 from field_names[b][t], else method_names[b-512][t]
__global__ void k_gather_names(const int* __restrict__ fn, const int* __restrict__ mn,
                               const float* __restrict__ tab, bf16* __restrict__ dst) {
  long long i = (long long)blockIdx.x * blockDim.x + threadIdx.x;
  if (i >= (long long)L_NAME * NB * E_DIM) return;
  int e = (int)(i % E_DIM);
  int tb = (int)(i / E_DIM);
  int b = tb % NB, t = tb / NB;
  int id = (b < NFLD) ? fn[b * L_NAME + t] : mn[(b - NFLD) * L_NAME + t];
  dst[i] = f2bf(tab[(size_t)id * E_DIM + e]);
}

// ---------------------------------------------------------------------------
// Register-blocked WMMA GEMM:
//   C[m,n] = sum_k A1[m,k]*W1[n,k] + sum_k A2[m,k]*W2[n,k] + b0[n] + b1[n]
// A row-major [M,K] bf16, W row-major [N,K] bf16 (torch Wih layout, B = W^T).
// Each wave computes a 64x64 macro-tile = 4x4 WMMA tiles (16 v8f accumulators),
// 4 A-frags + 4 B-frags -> 16 WMMAs per 32-wide k-step (1 b128 load per WMMA).
// Block = 256 threads = 8 waves arranged 2(M) x 4(N): block tile 128x256.
// Requires M % 128 == 0, N % 256 == 0, K % 32 == 0 (true for all call sites).
// ---------------------------------------------------------------------------
__device__ __forceinline__ v16bf load_a_frag(const bf16* __restrict__ base, int ld,
                                             int row0, int k0, int lane) {
  // 16x32 bf16 A tile: lanes 0-15 hold M=lane, K in [k0..k0+7] U [k0+16..k0+23];
  // lanes 16-31 hold M=lane-16, K in [k0+8..k0+15] U [k0+24..k0+31].
  int m  = row0 + (lane & 15);
  int hi = lane >> 4;
  const bf16* p = base + (size_t)m * ld + k0 + hi * 8;
  bf16x8 lo = *(const bf16x8*)p;
  bf16x8 up = *(const bf16x8*)(p + 16);
  v16bf r;
#pragma unroll
  for (int i = 0; i < 8; i++) { r[i] = lo[i]; r[8 + i] = up[i]; }
  return r;
}

__device__ __forceinline__ v16bf load_b_frag(const bf16* __restrict__ W, int ldw,
                                             int n0, int k0, int lane) {
  // 32x16 bf16 B tile: lanes 0-15 hold N=lane, K in [k0..k0+7] U [k0+16..k0+23]; lanes 16-31 K in [k0+8..k0+15] U [k0+24..k0+31].
  int n  = n0 + (lane & 15);
  int hi = lane >> 4;
  const bf16* p = W + (size_t)n * ldw + k0 + hi * 8;
  bf16x8 lo = *(const bf16x8*)p;
  bf16x8 up = *(const bf16x8*)(p + 16);
  v16bf r;
#pragma unroll
  for (int i = 0; i < 8; i++) { r[i] = lo[i]; r[8 + i] = up[i]; }
  return r;
}

__global__ __launch_bounds__(256) void k_wmma_gates(
    const bf16* __restrict__ A1, int K1, const bf16* __restrict__ W1,
    const bf16* __restrict__ A2, int K2, const bf16* __restrict__ W2,
    const float* __restrict__ b0, const float* __restrict__ b1,
    float* __restrict__ C, int N) {
  int lane = threadIdx.x & 31;
  int wave = threadIdx.x >> 5;
  int wm = wave >> 2;          // 0..1
  int wn = wave & 3;           // 0..3
  int m0 = blockIdx.y * 128 + wm * 64;
  int n0 = blockIdx.x * 256 + wn * 64;

  v8f acc[4][4];
#pragma unroll
  for (int i = 0; i < 4; i++)
#pragma unroll
    for (int j = 0; j < 4; j++) acc[i][j] = (v8f){};

  for (int k = 0; k < K1; k += 32) {
    v16bf a[4], b[4];
#pragma unroll
    for (int i = 0; i < 4; i++) a[i] = load_a_frag(A1, K1, m0 + 16 * i, k, lane);
#pragma unroll
    for (int j = 0; j < 4; j++) b[j] = load_b_frag(W1, K1, n0 + 16 * j, k, lane);
#pragma unroll
    for (int i = 0; i < 4; i++)
#pragma unroll
      for (int j = 0; j < 4; j++)
        acc[i][j] = __builtin_amdgcn_wmma_f32_16x16x32_bf16(false, a[i], false, b[j],
                                                            (short)0, acc[i][j], false, false);
  }
  if (A2) {
    for (int k = 0; k < K2; k += 32) {
      v16bf a[4], b[4];
#pragma unroll
      for (int i = 0; i < 4; i++) a[i] = load_a_frag(A2, K2, m0 + 16 * i, k, lane);
#pragma unroll
      for (int j = 0; j < 4; j++) b[j] = load_b_frag(W2, K2, n0 + 16 * j, k, lane);
#pragma unroll
      for (int i = 0; i < 4; i++)
#pragma unroll
        for (int j = 0; j < 4; j++)
          acc[i][j] = __builtin_amdgcn_wmma_f32_16x16x32_bf16(false, a[i], false, b[j],
                                                              (short)0, acc[i][j], false, false);
    }
  }
  // C layout per tile: VGPR v holds row m = v + 8*(lane>>4), col n = lane&15
  int hi = lane >> 4;
  int nl15 = lane & 15;
  float bias[4];
#pragma unroll
  for (int j = 0; j < 4; j++) {
    int n = n0 + 16 * j + nl15;
    bias[j] = b0[n] + b1[n];
  }
#pragma unroll
  for (int i = 0; i < 4; i++) {
#pragma unroll
    for (int j = 0; j < 4; j++) {
      int n = n0 + 16 * j + nl15;
#pragma unroll
      for (int v = 0; v < 8; v++) {
        int m = m0 + 16 * i + v + 8 * hi;
        C[(size_t)m * N + n] = acc[i][j][v] + bias[j];
      }
    }
  }
}

// ---------------------------------------------------------------------------
// Pointwise LSTM cell update (torch gate order i,f,g,o); h emitted in bf16
// so it feeds the next step's WMMA A-matrix directly.
// ---------------------------------------------------------------------------
__global__ void k_lstm_update(const float* __restrict__ gates, float* __restrict__ c,
                              bf16* __restrict__ h, int M) {
  int idx = blockIdx.x * blockDim.x + threadIdx.x;
  if (idx >= M * H_DIM) return;
  int b = idx >> 9, j = idx & (H_DIM - 1);
  const float* g = gates + (size_t)b * G4;
  float iv = sigm(g[j]);
  float fv = sigm(g[H_DIM + j]);
  float gv = tanhf(g[2 * H_DIM + j]);
  float ov = sigm(g[3 * H_DIM + j]);
  float cn = fv * c[idx] + iv * gv;
  c[idx] = cn;
  h[idx] = f2bf(ov * tanhf(cn));
}

// item-LSTM variant: also stores h (f32) into env[(item0+b)*2 + t][colOff + j]
__global__ void k_lstm_update_store(const float* __restrict__ gates, float* __restrict__ c,
                                    bf16* __restrict__ h, int M,
                                    float* __restrict__ env, int item0, int t, int colOff) {
  int idx = blockIdx.x * blockDim.x + threadIdx.x;
  if (idx >= M * H_DIM) return;
  int b = idx >> 9, j = idx & (H_DIM - 1);
  const float* g = gates + (size_t)b * G4;
  float iv = sigm(g[j]);
  float fv = sigm(g[H_DIM + j]);
  float gv = tanhf(g[2 * H_DIM + j]);
  float ov = sigm(g[3 * H_DIM + j]);
  float cn = fv * c[idx] + iv * gv;
  c[idx] = cn;
  float hn = ov * tanhf(cn);
  h[idx] = f2bf(hn);
  env[((size_t)(item0 + b) * 2 + t) * (2 * H_DIM) + colOff + j] = hn;
}

// copy name-LSTM final hidden [NB,512] bf16 into seq plane 1 at column colOff
__global__ void k_copy_hcat(const bf16* __restrict__ h, bf16* __restrict__ seq1, int colOff) {
  int idx = blockIdx.x * blockDim.x + threadIdx.x;
  if (idx >= NB * H_DIM) return;
  int b = idx >> 9, j = idx & (H_DIM - 1);
  seq1[(size_t)b * (2 * H_DIM) + colOff + j] = h[idx];
}

// ---------------------------------------------------------------------------
// Persistent batch-1 BiLSTM recurrence for nl (one workgroup per direction).
// Gin = x@Wih^T + bih + bhh precomputed for all t; here only h@Whh^T + cell.
// h staged in LDS (ds_load_b128 via float4), Whh stays L2-resident.
// ---------------------------------------------------------------------------
__global__ __launch_bounds__(512) void k_nl_recurrence(
    const float* __restrict__ Gin,   // [2][T_NL][G4]
    const float* __restrict__ WhhF, const float* __restrict__ WhhB,
    float* __restrict__ out) {       // [T_NL][1024]  fwd cols 0..511, bwd 512..1023
  int dir = blockIdx.x;
  const float* Whh = dir ? WhhB : WhhF;
  const float* G   = Gin + (size_t)dir * T_NL * G4;
  __shared__ float hs[H_DIM];
  int j = threadIdx.x;
  hs[j] = 0.0f;
  float c = 0.0f;
  const float* w_i = Whh + (size_t)j * H_DIM;
  const float* w_f = Whh + (size_t)(H_DIM + j) * H_DIM;
  const float* w_g = Whh + (size_t)(2 * H_DIM + j) * H_DIM;
  const float* w_o = Whh + (size_t)(3 * H_DIM + j) * H_DIM;
  __syncthreads();
  for (int s = 0; s < T_NL; s++) {
    int t = dir ? (T_NL - 1 - s) : s;
    const float* g = G + (size_t)t * G4;
    // prefetch next step's gate row while we chew on this one
    int tn = dir ? (t - 1) : (t + 1);
    if (tn >= 0 && tn < T_NL) __builtin_prefetch(G + (size_t)tn * G4 + j, 0, 1);
    float di = g[j], df = g[H_DIM + j], dg = g[2 * H_DIM + j], dq = g[3 * H_DIM + j];
    for (int k = 0; k < H_DIM; k += 4) {
      float4 h4 = *(const float4*)&hs[k];
      float4 a  = *(const float4*)&w_i[k];
      float4 b  = *(const float4*)&w_f[k];
      float4 cc = *(const float4*)&w_g[k];
      float4 d  = *(const float4*)&w_o[k];
      di += h4.x * a.x  + h4.y * a.y  + h4.z * a.z  + h4.w * a.w;
      df += h4.x * b.x  + h4.y * b.y  + h4.z * b.z  + h4.w * b.w;
      dg += h4.x * cc.x + h4.y * cc.y + h4.z * cc.z + h4.w * cc.w;
      dq += h4.x * d.x  + h4.y * d.y  + h4.z * d.z  + h4.w * d.w;
    }
    float iv = sigm(di), fv = sigm(df), gv = tanhf(dg), ov = sigm(dq);
    c = fv * c + iv * gv;
    float h = ov * tanhf(c);
    out[(size_t)t * (2 * H_DIM) + dir * H_DIM + j] = h;
    __syncthreads();
    hs[j] = h;
    __syncthreads();
  }
}

// ---------------------------------------------------------------------------
// Host: orchestrate the pipeline on `stream` (graph-capture safe).
// ---------------------------------------------------------------------------
static inline int ceil_div(long long a, int b) { return (int)((a + b - 1) / b); }

extern "C" void kernel_launch(void* const* d_in, const int* in_sizes, int n_in,
                              void* d_out, int out_size, void* d_ws, size_t ws_size,
                              hipStream_t stream) {
  (void)in_sizes; (void)n_in; (void)out_size; (void)ws_size;
  const int*   nl           = (const int*)d_in[0];
  const int*   field_types  = (const int*)d_in[1];
  const int*   field_names  = (const int*)d_in[2];
  const int*   method_types = (const int*)d_in[3];
  const int*   method_names = (const int*)d_in[4];
  const float* type_emb     = (const float*)d_in[5];
  const float* name_emb     = (const float*)d_in[6];
  // lstm leaves: fwd{Wih,Whh,bih,bhh} bwd{Wih,Whh,bih,bhh}
  const float* nlW[8]; const float* nmW[8]; const float* vaW[8]; const float* meW[8];
  for (int i = 0; i < 8; i++) {
    nlW[i] = (const float*)d_in[7 + i];
    nmW[i] = (const float*)d_in[15 + i];
    vaW[i] = (const float*)d_in[23 + i];
    meW[i] = (const float*)d_in[31 + i];
  }

  // ---- workspace layout ----
  char* base = (char*)d_ws;
  size_t off = 0;
  auto alloc = [&](size_t bytes) { void* p = base + off; off = (off + bytes + 255) & ~(size_t)255; return p; };
  bf16* nlX    = (bf16*)alloc((size_t)T_NL * E_DIM * 2);
  bf16* namesX = (bf16*)alloc((size_t)L_NAME * NB * E_DIM * 2);
  bf16* seq    = (bf16*)alloc((size_t)2 * NB * E_DIM * 2);           // plane0 types, plane1 h_name
  bf16* nlWih[2], *nmWih[2], *nmWhh[2], *vaWih[2], *vaWhh[2], *meWih[2], *meWhh[2];
  for (int d = 0; d < 2; d++) {
    nlWih[d] = (bf16*)alloc((size_t)G4 * E_DIM * 2);
    nmWih[d] = (bf16*)alloc((size_t)G4 * E_DIM * 2);
    nmWhh[d] = (bf16*)alloc((size_t)G4 * H_DIM * 2);
    vaWih[d] = (bf16*)alloc((size_t)G4 * E_DIM * 2);
    vaWhh[d] = (bf16*)alloc((size_t)G4 * H_DIM * 2);
    meWih[d] = (bf16*)alloc((size_t)G4 * E_DIM * 2);
    meWhh[d] = (bf16*)alloc((size_t)G4 * H_DIM * 2);
  }
  float* Gin   = (float*)alloc((size_t)2 * T_NL * G4 * 4);
  float* gates = (float*)alloc((size_t)NB * G4 * 4);
  float* cbuf  = (float*)alloc((size_t)NB * H_DIM * 4);
  bf16*  hbuf  = (bf16*)alloc((size_t)NB * H_DIM * 2);

  const int TPB = 256;
  auto conv = [&](const float* s, bf16* d, long long n) {
    k_f32_to_bf16<<<ceil_div(n, TPB), TPB, 0, stream>>>(s, d, (int)n);
  };
  // ---- weight conversion ----
  for (int d = 0; d < 2; d++) {
    conv(nlW[4 * d + 0], nlWih[d], (long long)G4 * E_DIM);
    conv(nmW[4 * d + 0], nmWih[d], (long long)G4 * E_DIM);
    conv(nmW[4 * d + 1], nmWhh[d], (long long)G4 * H_DIM);
    conv(vaW[4 * d + 0], vaWih[d], (long long)G4 * E_DIM);
    conv(vaW[4 * d + 1], vaWhh[d], (long long)G4 * H_DIM);
    conv(meW[4 * d + 0], meWih[d], (long long)G4 * E_DIM);
    conv(meW[4 * d + 1], meWhh[d], (long long)G4 * H_DIM);
  }
  // ---- gathers ----
  k_gather_rows<<<ceil_div((long long)T_NL * E_DIM, TPB), TPB, 0, stream>>>(
      nl, name_emb, nlX, T_NL, E_DIM);
  k_gather_names<<<ceil_div((long long)L_NAME * NB * E_DIM, TPB), TPB, 0, stream>>>(
      field_names, method_names, name_emb, namesX);
  k_gather_rows<<<ceil_div((long long)NFLD * E_DIM, TPB), TPB, 0, stream>>>(
      field_types, type_emb, seq, NFLD, E_DIM);
  k_gather_rows<<<ceil_div((long long)NFLD * E_DIM, TPB), TPB, 0, stream>>>(
      method_types, type_emb, seq + (size_t)NFLD * E_DIM, NFLD, E_DIM);

  // block tile = 128(M) x 256(N); wave tile = 64x64 (4x4 WMMA tiles)
  auto gemm = [&](const bf16* A1, int K1, const bf16* W1,
                  const bf16* A2, int K2, const bf16* W2,
                  const float* b0, const float* b1, float* C, int M, int N) {
    dim3 grid(N / 256, M / 128);
    k_wmma_gates<<<grid, 256, 0, stream>>>(A1, K1, W1, A2, K2, W2, b0, b1, C, N);
  };

  // ---- nl input projections (both directions), then persistent recurrence ----
  for (int d = 0; d < 2; d++)
    gemm(nlX, E_DIM, nlWih[d], nullptr, 0, nullptr,
         nlW[4 * d + 2], nlW[4 * d + 3], Gin + (size_t)d * T_NL * G4, T_NL, G4);
  k_nl_recurrence<<<2, H_DIM, 0, stream>>>(Gin, nlW[1], nlW[5], (float*)d_out);

  // ---- name BiLSTM: 32 steps, batch 1024, WMMA per step ----
  for (int d = 0; d < 2; d++) {
    hipMemsetAsync(cbuf, 0, (size_t)NB * H_DIM * 4, stream);
    hipMemsetAsync(hbuf, 0, (size_t)NB * H_DIM * 2, stream);
    for (int s = 0; s < L_NAME; s++) {
      int t = d ? (L_NAME - 1 - s) : s;
      gemm(namesX + (size_t)t * NB * E_DIM, E_DIM, nmWih[d],
           hbuf, H_DIM, nmWhh[d],
           nmW[4 * d + 2], nmW[4 * d + 3], gates, NB, G4);
      k_lstm_update<<<ceil_div((long long)NB * H_DIM, TPB), TPB, 0, stream>>>(gates, cbuf, hbuf, NB);
    }
    k_copy_hcat<<<ceil_div((long long)NB * H_DIM, TPB), TPB, 0, stream>>>(
        hbuf, seq + (size_t)NB * E_DIM, d * H_DIM);
  }

  // ---- item BiLSTMs: seq=[types, h_name], T=2; var items 0..511, method 512..1023 ----
  float* env = (float*)d_out + (size_t)T_NL * (2 * H_DIM);
  for (int set = 0; set < 2; set++) {
    const bf16* const* Wih = set ? (const bf16* const*)meWih : (const bf16* const*)vaWih;
    const bf16* const* Whh = set ? (const bf16* const*)meWhh : (const bf16* const*)vaWhh;
    const float* const* Wb = set ? meW : vaW;
    int item0 = set * NFLD;
    for (int d = 0; d < 2; d++) {
      hipMemsetAsync(cbuf, 0, (size_t)NFLD * H_DIM * 4, stream);
      hipMemsetAsync(hbuf, 0, (size_t)NFLD * H_DIM * 2, stream);
      for (int s = 0; s < 2; s++) {
        int t = d ? (1 - s) : s;
        gemm(seq + ((size_t)t * NB + item0) * E_DIM, E_DIM, Wih[d],
             hbuf, H_DIM, Whh[d],
             Wb[4 * d + 2], Wb[4 * d + 3], gates, NFLD, G4);
        k_lstm_update_store<<<ceil_div((long long)NFLD * H_DIM, TPB), TPB, 0, stream>>>(
            gates, cbuf, hbuf, NFLD, env, item0, t, d * H_DIM);
      }
    }
  }
}
